// VGRAMLayer_31310311587964
// MI455X (gfx1250) — matrix operand order repaired and checked by reading, not compiled, not run on hardware
//
#include <hip/hip_runtime.h>
#include <hip/hip_fp16.h>

typedef _Float16 half16 __attribute__((ext_vector_type(16)));
typedef _Float16 half8  __attribute__((ext_vector_type(8)));
typedef float    float8 __attribute__((ext_vector_type(8)));

#define B_  32
#define IN_ 4096
#define N_  2048
#define P_  128
#define M_  256
#define D_  10

// sigmoid(t) = 0.5*tanh(t/2) + 0.5 : single TRANS op (v_tanh_f32) when available,
// else exp+rcp (2 TRANS ops, still no IEEE div expansion).
__device__ __forceinline__ float sigmoidf_(float t) {
#if __has_builtin(__builtin_amdgcn_tanhf)
    return __builtin_fmaf(__builtin_amdgcn_tanhf(t * 0.5f), 0.5f, 0.5f);
#else
    float e = __expf(-t);
    return __builtin_amdgcn_rcpf(1.0f + e);
#endif
}

__device__ __forceinline__ float fast_sqrtf_(float t) {
#if __has_builtin(__builtin_amdgcn_sqrtf)
    return __builtin_amdgcn_sqrtf(t);   // raw v_sqrt_f32, no range fixup
#else
    return sqrtf(t);
#endif
}

__global__ __launch_bounds__(256)
void vgram_fused_kernel(const float* __restrict__ x,
                        const int*   __restrict__ idx_a,
                        const int*   __restrict__ idx_b,
                        const float* __restrict__ keys_raw,
                        const float* __restrict__ values,
                        float*       __restrict__ out)
{
    const int n    = blockIdx.x;      // neuron
    const int t    = threadIdx.x;     // 0..255
    const int lane = t & 31;
    const int wave = t >> 5;          // 0..7

    __shared__ int      s_ia[P_];
    __shared__ int      s_ib[P_];
    __shared__ _Float16 s_bits[B_][136];        // 32 x 128 f16, padded rows (conflict-free A-frag loads)
    __shared__ float    s_bsq[B_];
    __shared__ float    s_vals[M_ * 11];        // 256 x 10 padded to 11
    __shared__ float    s_dist[B_][257];        // 32 x 256 padded

    // ---- stage idx pairs, values, zero b_sq ----
    if (t < P_) { s_ia[t] = idx_a[n * P_ + t]; s_ib[t] = idx_b[n * P_ + t]; }
    if (t < B_) s_bsq[t] = 0.0f;
    {
        const float* vsrc = values + (size_t)n * (M_ * D_);
        for (int i = t; i < M_ * D_; i += 256) {
            int m = i / D_, d = i - m * D_;
            s_vals[m * 11 + d] = vsrc[i];
        }
    }
    __syncthreads();

    // ---- Minchinton bits: sigmoid of gathered differences; accumulate b_sq ----
    for (int i = t; i < B_ * P_; i += 256) {
        int b = i >> 7, p = i & 127;
        float v = sigmoidf_(x[b * IN_ + s_ia[p]] - x[b * IN_ + s_ib[p]]);
        s_bits[b][p] = (_Float16)v;
        atomicAdd(&s_bsq[b], v * v);
    }
    __syncthreads();

    // ---- preload A fragments (contiguous k order, same as B frag) ----
    // lane<16: M=lane, K = kb..kb+15 with kb=0
    // lane>=16: M=lane-16, kb=16
    half16 afrag[2][4];
    {
        const int mrow = lane & 15;
        const int kb   = (lane < 16) ? 0 : 16;
        #pragma unroll
        for (int mt = 0; mt < 2; ++mt) {
            const _Float16* row = &s_bits[mt * 16 + mrow][0];
            #pragma unroll
            for (int ks = 0; ks < 4; ++ks) {
                union { half16 v; half8 h[2]; } u;
                u.h[0] = *(const half8*)(row + ks * 32 + kb);
                u.h[1] = *(const half8*)(row + ks * 32 + kb + 8);
                afrag[mt][ks] = u.v;
            }
        }
    }

    // ---- GEMM1: dots = bits(32x128) x keys^T(128x256), keys streamed once ----
    const float* kptr_base = keys_raw + (size_t)n * (M_ * P_);
    const int ncol = lane & 15;
    const int kb2  = (lane < 16) ? 0 : 16;   // B frag: lane<16 holds K 0..15, lane>=16 holds K 16..31
    const int rowoff = (lane < 16) ? 0 : 8;  // C/D frag row offset

    #pragma unroll
    for (int nti = 0; nti < 2; ++nti) {
        const int nt = wave * 2 + nti;       // this wave's N-tile (both M-tiles -> keys read once)
        const int m  = nt * 16 + ncol;       // memory-entry index this lane's column holds
        float8 acc0 = {0.f,0.f,0.f,0.f,0.f,0.f,0.f,0.f};
        float8 acc1 = {0.f,0.f,0.f,0.f,0.f,0.f,0.f,0.f};
        float  ksqa = 0.0f, ksqb = 0.0f;     // two accumulators -> v_pk_fma_f32 friendly

        #pragma unroll
        for (int ks = 0; ks < 4; ++ks) {
            const float* kp = kptr_base + m * P_ + ks * 32 + kb2;
            union { half16 v; _Float16 e[16]; } bu;
            #pragma unroll
            for (int j = 0; j < 4; ++j) {
                float4 f = *(const float4*)(kp + j * 4);
                float s0 = sigmoidf_(f.x), s1 = sigmoidf_(f.y);
                float s2 = sigmoidf_(f.z), s3 = sigmoidf_(f.w);
                ksqa += s0 * s0 + s2 * s2;
                ksqb += s1 * s1 + s3 * s3;
                bu.e[j*4+0] = (_Float16)s0; bu.e[j*4+1] = (_Float16)s1;
                bu.e[j*4+2] = (_Float16)s2; bu.e[j*4+3] = (_Float16)s3;
            }
            acc0 = __builtin_amdgcn_wmma_f32_16x16x32_f16(
                       false, afrag[0][ks], false, bu.v, (short)0, acc0, false, false);
            acc1 = __builtin_amdgcn_wmma_f32_16x16x32_f16(
                       false, afrag[1][ks], false, bu.v, (short)0, acc1, false, false);
        }
        // combine the two K-halves of k_sq (lane l and l^16 hold same m)
        float ksq = ksqa + ksqb;
        float ksq_full = ksq + __shfl_xor(ksq, 16, 32);

        // distances -> LDS
        #pragma unroll
        for (int r = 0; r < 8; ++r) {
            int b0 = r + rowoff;            // M-tile 0
            float d20 = s_bsq[b0] + ksq_full - 2.0f * acc0[r];
            s_dist[b0][m] = fast_sqrtf_(fmaxf(d20, 0.0f) + 1e-12f);
            int b1 = 16 + r + rowoff;       // M-tile 1
            float d21 = s_bsq[b1] + ksq_full - 2.0f * acc1[r];
            s_dist[b1][m] = fast_sqrtf_(fmaxf(d21, 0.0f) + 1e-12f);
        }
    }
    __syncthreads();

    // ---- softmin over M + weighted retrieval of values (per wave: 4 batch rows) ----
    #pragma unroll 1
    for (int br = 0; br < 4; ++br) {
        const int b = wave * 4 + br;

        float dmin = 1e30f;
        #pragma unroll
        for (int j = 0; j < 8; ++j) dmin = fminf(dmin, s_dist[b][lane + 32 * j]);
        #pragma unroll
        for (int off = 16; off > 0; off >>= 1) dmin = fminf(dmin, __shfl_xor(dmin, off, 32));

        float sum = 0.0f;
        float oacc[D_];
        #pragma unroll
        for (int d = 0; d < D_; ++d) oacc[d] = 0.0f;

        #pragma unroll
        for (int j = 0; j < 8; ++j) {
            int mm = lane + 32 * j;
            float w = __expf(dmin - s_dist[b][mm]);   // BETA = 1
            sum += w;
            const float* vrow = &s_vals[mm * 11];
            #pragma unroll
            for (int d = 0; d < D_; ++d) oacc[d] += w * vrow[d];
        }
        #pragma unroll
        for (int off = 16; off > 0; off >>= 1) {
            sum += __shfl_xor(sum, off, 32);
            #pragma unroll
            for (int d = 0; d < D_; ++d) oacc[d] += __shfl_xor(oacc[d], off, 32);
        }
        if (lane == 0) {
            float inv = __builtin_amdgcn_rcpf(sum);   // v_rcp_f32, not IEEE div
            float* orow = out + ((size_t)b * N_ + n) * D_;
            #pragma unroll
            for (int d = 0; d < D_; ++d) orow[d] = oacc[d] * inv;
        }
    }
}

extern "C" void kernel_launch(void* const* d_in, const int* in_sizes, int n_in,
                              void* d_out, int out_size, void* d_ws, size_t ws_size,
                              hipStream_t stream) {
    const float* x        = (const float*)d_in[0];
    const int*   idx_a    = (const int*)  d_in[1];
    const int*   idx_b    = (const int*)  d_in[2];
    const float* keys_raw = (const float*)d_in[3];
    const float* values   = (const float*)d_in[4];
    float*       out      = (float*)d_out;

    vgram_fused_kernel<<<N_, 256, 0, stream>>>(x, idx_a, idx_b, keys_raw, values, out);
}
